// MultiHeadBlock_43052752175768
// MI455X (gfx1250) — matrix operand order repaired and checked by reading, not compiled, not run on hardware
//
#include <hip/hip_runtime.h>
#include <math.h>

// ---------------------------------------------------------------------------
// MultiHeadBlock on gfx1250 (MI455X).
//  - One-time f32->bf16 conversion / transposition of all operands so every
//    GEMM is bf16 A[M,K] x B^T (B stored [N,K]) with f32 WMMA accumulation.
//  - GEMM: 128x64 block tile, BK=64, 8 wave32 waves, 32x32 accum per wave
//    (8 v_wmma_f32_16x16x32_bf16 per wave per K-step).
//  - Double-buffered LDS fed by GLOBAL_LOAD_ASYNC_TO_LDS_B128 (ASYNCcnt path)
//    when the builtin exists; synchronous uint4 staging otherwise.
// ---------------------------------------------------------------------------

#define SEQ    4096
#define DIM    1024
#define HEADS  8
#define CONCAT (HEADS * DIM)   // 8192
#define LN_EPS 1e-5f

#if defined(__gfx1250__) && \
    __has_builtin(__builtin_amdgcn_global_load_async_to_lds_b128) && \
    __has_builtin(__builtin_amdgcn_s_wait_asynccnt)
#define ASYNC_COPY 1
#else
#define ASYNC_COPY 0
#endif

typedef __attribute__((ext_vector_type(16))) __bf16        v16bf;
typedef __attribute__((ext_vector_type(8)))  float         v8f;
typedef __attribute__((ext_vector_type(8)))  unsigned int  v8u;
typedef int v4i_vec __attribute__((vector_size(4 * sizeof(int))));

__device__ __forceinline__ unsigned short f2bf(float f) {
  unsigned int u = __float_as_uint(f);
  u += 0x7FFFu + ((u >> 16) & 1u);          // round-to-nearest-even
  return (unsigned short)(u >> 16);
}

__device__ __forceinline__ void copy16_async(const unsigned short* g, unsigned short* lds) {
#if ASYNC_COPY
  __builtin_amdgcn_global_load_async_to_lds_b128(
      (__attribute__((address_space(1))) v4i_vec*)g,
      (__attribute__((address_space(3))) v4i_vec*)lds, 0, 0);
#else
  *(uint4*)lds = *(const uint4*)g;
#endif
}

__device__ __forceinline__ void wait_async_all() {
#if ASYNC_COPY
  __builtin_amdgcn_s_wait_asynccnt(0);
#endif
}

__device__ __forceinline__ v16bf make_frag(const unsigned short* p0,
                                           const unsigned short* p1) {
  const uint4 x0 = *(const uint4*)p0;
  const uint4 x1 = *(const uint4*)p1;
  v8u u;
  u[0] = x0.x; u[1] = x0.y; u[2] = x0.z; u[3] = x0.w;
  u[4] = x1.x; u[5] = x1.y; u[6] = x1.z; u[7] = x1.w;
  return __builtin_bit_cast(v16bf, u);
}

#define WMMA_BF16(A, B, C) \
  __builtin_amdgcn_wmma_f32_16x16x32_bf16(false, (A), false, (B), (short)0, (C), false, false)

// ---------------------------------------------------------------------------
// GEMM: C[M,N] = act( alpha * A[M,K] x B[N,K]^T + bias )
//   A: bf16 row-major [M,K] (lda); B: bf16 row-major [N,K] (ldb)
//   OUT_TRANS: write C transposed, i.e. Ct[n*ldc + m]
// Requirements: M % 128 == 0, N % 64 == 0, K % 64 == 0 (true for all calls).
// Block: 256 threads (8 wave32 waves), tile 128x64, K-step 64.
// ---------------------------------------------------------------------------
template<bool OUT_BF16, bool RELU, bool OUT_TRANS>
__global__ __launch_bounds__(256, 2)
void gemm_wmma_kernel(const unsigned short* __restrict__ Ap,
                      const unsigned short* __restrict__ Bp,
                      void* __restrict__ Cp, const float* __restrict__ bias,
                      int K, int lda, int ldb, int ldc, float alpha)
{
  constexpr int BM = 128, BN = 64, BK = 64;
  constexpr int PITCH = BK + 16;                 // 80 bf16 -> 160B rows (16B aligned)
  __shared__ unsigned short As[2][BM * PITCH];
  __shared__ unsigned short Bs[2][BN * PITCH];

  const int tid  = threadIdx.x;
  const int lane = tid & 31;
  const int wave = tid >> 5;
  const int m0   = (wave >> 1) * 32;             // 0,32,64,96
  const int n0   = (wave & 1) * 32;              // 0,32
  const int bm   = blockIdx.y * BM;
  const int bn   = blockIdx.x * BN;
  const int half = lane >> 4;
  const int l15  = lane & 15;

  v8f acc00 = {0.f,0.f,0.f,0.f,0.f,0.f,0.f,0.f};
  v8f acc01 = acc00, acc10 = acc00, acc11 = acc00;

  // staging decomposition: 16B (8 bf16) chunks
  //   A tile: 128 rows x 8 chunks = 1024 chunks -> 4 per thread
  //   B tile:  64 rows x 8 chunks =  512 chunks -> 2 per thread
  const int arow0 = tid >> 3;            // base rows for chunk 0 (+32 per i)
  const int kk8   = (tid & 7) * 8;

  #define STAGE(buf, k0)                                                        \
    {                                                                           \
      _Pragma("unroll")                                                         \
      for (int i = 0; i < 4; ++i) {                                             \
        const int row = arow0 + i * 32;                                         \
        copy16_async(Ap + (size_t)(bm + row) * lda + (k0) + kk8,                \
                     &As[buf][row * PITCH + kk8]);                              \
      }                                                                         \
      _Pragma("unroll")                                                         \
      for (int i = 0; i < 2; ++i) {                                             \
        const int row = arow0 + i * 32;                                         \
        copy16_async(Bp + (size_t)(bn + row) * ldb + (k0) + kk8,                \
                     &Bs[buf][row * PITCH + kk8]);                              \
      }                                                                         \
    }

#if ASYNC_COPY
  STAGE(0, 0);
  wait_async_all();
  __syncthreads();
  for (int k0 = 0; k0 < K; k0 += BK) {
    const int cur = (k0 >> 6) & 1;
    if (k0 + BK < K) STAGE(cur ^ 1, k0 + BK);
#else
  STAGE(0, 0);
  for (int k0 = 0; k0 < K; k0 += BK) {
    const int cur = (k0 >> 6) & 1;
    __syncthreads();                    // staged tile visible to all waves
    if (k0 + BK < K) STAGE(cur ^ 1, k0 + BK);
#endif

    const unsigned short* a0p = &As[cur][(m0 +  0 + l15) * PITCH];
    const unsigned short* a1p = &As[cur][(m0 + 16 + l15) * PITCH];
    const unsigned short* b0p = &Bs[cur][(n0 +  0 + l15) * PITCH];
    const unsigned short* b1p = &Bs[cur][(n0 + 16 + l15) * PITCH];

    #pragma unroll
    for (int ks = 0; ks < BK; ks += 32) {
      // A 16x32 frag: uint4 @ ks + half*8 and @ ks + 16 + half*8
      const v16bf a0 = make_frag(a0p + ks + half * 8, a0p + ks + 16 + half * 8);
      const v16bf a1 = make_frag(a1p + ks + half * 8, a1p + ks + 16 + half * 8);
      // B 32x16 frag: uint4 @ ks + half*8 and @ ks + 16 + half*8
      const v16bf b0 = make_frag(b0p + ks + half * 8, b0p + ks + 16 + half * 8);
      const v16bf b1 = make_frag(b1p + ks + half * 8, b1p + ks + 16 + half * 8);

      acc00 = WMMA_BF16(a0, b0, acc00);
      acc01 = WMMA_BF16(a0, b1, acc01);
      acc10 = WMMA_BF16(a1, b0, acc10);
      acc11 = WMMA_BF16(a1, b1, acc11);
    }

#if ASYNC_COPY
    wait_async_all();                   // our prefetch of buf^1 landed
    __syncthreads();                    // everyone done reading cur + prefetching
#endif
  }
  #undef STAGE

  // epilogue: C/D layout: lane(l15,half), reg r -> M = half*8 + r, N = l15
  const int cn0 = bn + n0 + l15;
  const int cn1 = bn + n0 + 16 + l15;
  const float bv0 = (!OUT_TRANS && bias) ? bias[cn0] : 0.f;
  const float bv1 = (!OUT_TRANS && bias) ? bias[cn1] : 0.f;

  #pragma unroll
  for (int mt = 0; mt < 2; ++mt) {
    const v8f* accA = mt ? &acc10 : &acc00;
    const v8f* accB = mt ? &acc11 : &acc01;
    #pragma unroll
    for (int r = 0; r < 8; ++r) {
      const int cm = bm + m0 + mt * 16 + half * 8 + r;
      float v0 = (*accA)[r] * alpha + bv0;
      float v1 = (*accB)[r] * alpha + bv1;
      if (RELU) { v0 = fmaxf(v0, 0.f); v1 = fmaxf(v1, 0.f); }
      if (OUT_TRANS) {
        unsigned short* C = (unsigned short*)Cp;
        C[(size_t)cn0 * ldc + cm] = f2bf(v0);
        C[(size_t)cn1 * ldc + cm] = f2bf(v1);
      } else if (OUT_BF16) {
        unsigned short* C = (unsigned short*)Cp;
        C[(size_t)cm * ldc + cn0] = f2bf(v0);
        C[(size_t)cm * ldc + cn1] = f2bf(v1);
      } else {
        float* C = (float*)Cp;
        C[(size_t)cm * ldc + cn0] = v0;
        C[(size_t)cm * ldc + cn1] = v1;
      }
    }
  }
}

// ---------------------------------------------------------------------------
// f32 -> bf16 elementwise convert (n must be a multiple of 4)
// ---------------------------------------------------------------------------
__global__ __launch_bounds__(256)
void cvt_bf16_kernel(const float* __restrict__ in, unsigned short* __restrict__ out,
                     int n4)
{
  const int i = blockIdx.x * 256 + threadIdx.x;
  if (i < n4) {
    const float4 v = ((const float4*)in)[i];
    ushort4 o;
    o.x = f2bf(v.x); o.y = f2bf(v.y); o.z = f2bf(v.z); o.w = f2bf(v.w);
    ((ushort4*)out)[i] = o;
  }
}

// ---------------------------------------------------------------------------
// Batched f32 [K,N] -> bf16 [N,K] transpose-convert (32x32 LDS tiles).
// grid = (N/32, K/32, batch); block = 256 (32x8).
// ---------------------------------------------------------------------------
__global__ __launch_bounds__(256)
void tcvt_bf16_kernel(const float* __restrict__ W, unsigned short* __restrict__ Wt,
                      int K, int N)
{
  __shared__ float t[32][33];
  const size_t boff = (size_t)blockIdx.z * K * N;
  const float* w = W + boff;
  unsigned short* wt = Wt + boff;
  const int nb = blockIdx.x * 32, kb = blockIdx.y * 32;
  const int tx = threadIdx.x & 31, ty = threadIdx.x >> 5;   // ty: 0..7
  #pragma unroll
  for (int j = 0; j < 4; ++j) {
    const int r = ty + j * 8;
    t[r][tx] = w[(size_t)(kb + r) * N + nb + tx];
  }
  __syncthreads();
  #pragma unroll
  for (int j = 0; j < 4; ++j) {
    const int r = ty + j * 8;
    wt[(size_t)(nb + r) * K + kb + tx] = f2bf(t[tx][r]);
  }
}

// ---------------------------------------------------------------------------
// Row softmax over N columns: f32 in -> bf16 probabilities out. 1 block/row.
// ---------------------------------------------------------------------------
__global__ __launch_bounds__(256)
void softmax_rows_kernel(const float* __restrict__ S,
                         unsigned short* __restrict__ P, int N)
{
  __shared__ float red[256];
  const int row = blockIdx.x;
  const int tid = threadIdx.x;
  const float* s = S + (size_t)row * N;

  float m = -3.402823466e38f;
  for (int c = tid; c < N; c += 256) m = fmaxf(m, s[c]);
  red[tid] = m; __syncthreads();
  for (int st = 128; st > 0; st >>= 1) {
    if (tid < st) red[tid] = fmaxf(red[tid], red[tid + st]);
    __syncthreads();
  }
  m = red[0]; __syncthreads();

  float sum = 0.f;
  for (int c = tid; c < N; c += 256) sum += __expf(s[c] - m);
  red[tid] = sum; __syncthreads();
  for (int st = 128; st > 0; st >>= 1) {
    if (tid < st) red[tid] += red[tid + st];
    __syncthreads();
  }
  const float inv = 1.0f / red[0];

  unsigned short* p = P + (size_t)row * N;
  for (int c = tid; c < N; c += 256) p[c] = f2bf(__expf(s[c] - m) * inv);
}

// ---------------------------------------------------------------------------
// LN1: x1 = LN(attn + decoder) * g + b ; emit f32 and bf16 copies.
// ---------------------------------------------------------------------------
__global__ __launch_bounds__(256)
void ln1_kernel(const float* __restrict__ attn, const float* __restrict__ dec,
                const float* __restrict__ g, const float* __restrict__ b,
                float* __restrict__ x1f, unsigned short* __restrict__ x1b)
{
  __shared__ float red[256];
  const int row = blockIdx.x, tid = threadIdx.x;
  float v[4];
  float sum = 0.f;
  #pragma unroll
  for (int j = 0; j < 4; ++j) {
    const int c = tid + j * 256;
    v[j] = attn[(size_t)row * DIM + c] + dec[(size_t)row * DIM + c];
    sum += v[j];
  }
  red[tid] = sum; __syncthreads();
  for (int st = 128; st > 0; st >>= 1) { if (tid < st) red[tid] += red[tid + st]; __syncthreads(); }
  const float mu = red[0] * (1.0f / DIM);
  __syncthreads();
  float sq = 0.f;
  #pragma unroll
  for (int j = 0; j < 4; ++j) { const float d = v[j] - mu; sq += d * d; }
  red[tid] = sq; __syncthreads();
  for (int st = 128; st > 0; st >>= 1) { if (tid < st) red[tid] += red[tid + st]; __syncthreads(); }
  const float rstd = rsqrtf(red[0] * (1.0f / DIM) + LN_EPS);
  #pragma unroll
  for (int j = 0; j < 4; ++j) {
    const int c = tid + j * 256;
    const float y = (v[j] - mu) * rstd * g[c] + b[c];
    x1f[(size_t)row * DIM + c] = y;
    x1b[(size_t)row * DIM + c] = f2bf(y);
  }
}

// ---------------------------------------------------------------------------
// LN2 + final residual: out = LN(adj + x1) * g + b + decoder
// ---------------------------------------------------------------------------
__global__ __launch_bounds__(256)
void ln2_out_kernel(const float* __restrict__ adj, const float* __restrict__ x1,
                    const float* __restrict__ g, const float* __restrict__ b,
                    const float* __restrict__ dec, float* __restrict__ out)
{
  __shared__ float red[256];
  const int row = blockIdx.x, tid = threadIdx.x;
  float v[4];
  float sum = 0.f;
  #pragma unroll
  for (int j = 0; j < 4; ++j) {
    const int c = tid + j * 256;
    v[j] = adj[(size_t)row * DIM + c] + x1[(size_t)row * DIM + c];
    sum += v[j];
  }
  red[tid] = sum; __syncthreads();
  for (int st = 128; st > 0; st >>= 1) { if (tid < st) red[tid] += red[tid + st]; __syncthreads(); }
  const float mu = red[0] * (1.0f / DIM);
  __syncthreads();
  float sq = 0.f;
  #pragma unroll
  for (int j = 0; j < 4; ++j) { const float d = v[j] - mu; sq += d * d; }
  red[tid] = sq; __syncthreads();
  for (int st = 128; st > 0; st >>= 1) { if (tid < st) red[tid] += red[tid + st]; __syncthreads(); }
  const float rstd = rsqrtf(red[0] * (1.0f / DIM) + LN_EPS);
  #pragma unroll
  for (int j = 0; j < 4; ++j) {
    const int c = tid + j * 256;
    out[(size_t)row * DIM + c] =
        (v[j] - mu) * rstd * g[c] + b[c] + dec[(size_t)row * DIM + c];
  }
}

// ---------------------------------------------------------------------------

static inline dim3 gemm_grid(int M, int N) { return dim3(N / 64, M / 128); }

extern "C" void kernel_launch(void* const* d_in, const int* in_sizes, int n_in,
                              void* d_out, int out_size, void* d_ws, size_t ws_size,
                              hipStream_t stream)
{
  (void)in_sizes; (void)n_in; (void)out_size; (void)ws_size;

  const float* enc   = (const float*)d_in[0];   // [4096,1024]
  const float* dec   = (const float*)d_in[1];   // [4096,1024]
  const float* WQ    = (const float*)d_in[2];   // [8,1024,1024] (in,out)
  const float* WK    = (const float*)d_in[3];
  const float* WV    = (const float*)d_in[4];
  const float* WO_w  = (const float*)d_in[5];   // [1024,8192]  (out,in) = [N,K]
  const float* WO_b  = (const float*)d_in[6];
  const float* ln1_g = (const float*)d_in[7];
  const float* ln1_b = (const float*)d_in[8];
  const float* FF_w  = (const float*)d_in[9];   // [1024,1024]  (out,in) = [N,K]
  const float* FF_b  = (const float*)d_in[10];
  const float* FF2_w = (const float*)d_in[11];
  const float* FF2_b = (const float*)d_in[12];
  const float* ln2_g = (const float*)d_in[13];
  const float* ln2_b = (const float*)d_in[14];

  // ---- workspace carve-up (~244 MB) ----
  char* ws = (char*)d_ws;
  size_t off = 0;
  #define CARVE(ty, name, elems) ty* name = (ty*)(ws + off); off += (size_t)(elems) * sizeof(ty)
  CARVE(unsigned short, enc_b,  (size_t)SEQ * DIM);           // 8 MB
  CARVE(unsigned short, dec_b,  (size_t)SEQ * DIM);           // 8 MB
  CARVE(unsigned short, WQt,    (size_t)HEADS * DIM * DIM);   // 16 MB  [h][out N][in K]
  CARVE(unsigned short, WKt,    (size_t)HEADS * DIM * DIM);   // 16 MB
  CARVE(unsigned short, WVt,    (size_t)HEADS * DIM * DIM);   // 16 MB
  CARVE(unsigned short, WOb,    (size_t)DIM * CONCAT);        // 16 MB  [N,K] already
  CARVE(unsigned short, FFwb,   (size_t)DIM * DIM);           // 2 MB
  CARVE(unsigned short, FF2wb,  (size_t)DIM * DIM);           // 2 MB
  CARVE(unsigned short, Qh,     (size_t)SEQ * DIM);           // 8 MB   [Sq,K]
  CARVE(unsigned short, Kh,     (size_t)SEQ * DIM);           // 8 MB   [Sk,K] == [N,K] for scores
  CARVE(unsigned short, Vt,     (size_t)DIM * SEQ);           // 8 MB   [A,Sk] == [N,K] for P@V
  CARVE(float,          Sbuf,   (size_t)1024 * SEQ);          // 16 MB
  CARVE(unsigned short, Pbuf,   (size_t)1024 * SEQ);          // 8 MB
  CARVE(unsigned short, concat, (size_t)SEQ * CONCAT);        // 64 MB
  CARVE(float,          tmpf,   (size_t)SEQ * DIM);           // 16 MB
  CARVE(float,          x1f,    (size_t)SEQ * DIM);           // 16 MB
  CARVE(unsigned short, x1b,    (size_t)SEQ * DIM);           // 8 MB
  CARVE(unsigned short, hb,     (size_t)SEQ * DIM);           // 8 MB
  #undef CARVE

  const float scale = 1.0f / 32.0f;   // 1/sqrt(1024)
  const int   QCH   = 1024;

  // ---- one-time dtype conversions ----
  {
    const int n4a = SEQ * DIM / 4;
    cvt_bf16_kernel<<<(n4a + 255) / 256, 256, 0, stream>>>(enc, enc_b, n4a);
    cvt_bf16_kernel<<<(n4a + 255) / 256, 256, 0, stream>>>(dec, dec_b, n4a);
    dim3 tg(DIM / 32, DIM / 32, HEADS);
    tcvt_bf16_kernel<<<tg, 256, 0, stream>>>(WQ, WQt, DIM, DIM);
    tcvt_bf16_kernel<<<tg, 256, 0, stream>>>(WK, WKt, DIM, DIM);
    tcvt_bf16_kernel<<<tg, 256, 0, stream>>>(WV, WVt, DIM, DIM);
    const int n4o = DIM * CONCAT / 4;
    cvt_bf16_kernel<<<(n4o + 255) / 256, 256, 0, stream>>>(WO_w, WOb, n4o);
    const int n4f = DIM * DIM / 4;
    cvt_bf16_kernel<<<(n4f + 255) / 256, 256, 0, stream>>>(FF_w, FFwb, n4f);
    cvt_bf16_kernel<<<(n4f + 255) / 256, 256, 0, stream>>>(FF2_w, FF2wb, n4f);
  }

  for (int h = 0; h < HEADS; ++h) {
    const unsigned short* wq = WQt + (size_t)h * DIM * DIM;
    const unsigned short* wk = WKt + (size_t)h * DIM * DIM;
    const unsigned short* wv = WVt + (size_t)h * DIM * DIM;

    // Q_h = dec @ WQ_h ; K_h = enc @ WK_h (bf16); V_h^T (bf16, transposed out)
    gemm_wmma_kernel<true, false, false><<<gemm_grid(SEQ, DIM), 256, 0, stream>>>(
        dec_b, wq, Qh, nullptr, DIM, DIM, DIM, DIM, 1.0f);
    gemm_wmma_kernel<true, false, false><<<gemm_grid(SEQ, DIM), 256, 0, stream>>>(
        enc_b, wk, Kh, nullptr, DIM, DIM, DIM, DIM, 1.0f);
    gemm_wmma_kernel<true, false, true><<<gemm_grid(SEQ, DIM), 256, 0, stream>>>(
        enc_b, wv, Vt, nullptr, DIM, DIM, DIM, SEQ, 1.0f);

    for (int q0 = 0; q0 < SEQ; q0 += QCH) {
      // S = (Q chunk) @ K_h^T * scale   (f32)
      gemm_wmma_kernel<false, false, false><<<gemm_grid(QCH, SEQ), 256, 0, stream>>>(
          Qh + (size_t)q0 * DIM, Kh, Sbuf, nullptr, DIM, DIM, DIM, SEQ, scale);
      // row softmax -> bf16 probabilities
      softmax_rows_kernel<<<QCH, 256, 0, stream>>>(Sbuf, Pbuf, SEQ);
      // attn chunk = P @ V_h (B = Vt [A,Sk]) -> concat[:, h*1024..]
      gemm_wmma_kernel<true, false, false><<<gemm_grid(QCH, DIM), 256, 0, stream>>>(
          Pbuf, Vt, concat + (size_t)q0 * CONCAT + (size_t)h * DIM, nullptr,
          SEQ, SEQ, SEQ, CONCAT, 1.0f);
    }
  }

  // attention = concat @ WO_w^T + WO_b   (f32)
  gemm_wmma_kernel<false, false, false><<<gemm_grid(SEQ, DIM), 256, 0, stream>>>(
      concat, WOb, tmpf, WO_b, CONCAT, CONCAT, CONCAT, DIM, 1.0f);
  // x1 = LN(attention + decoder)
  ln1_kernel<<<SEQ, 256, 0, stream>>>(tmpf, dec, ln1_g, ln1_b, x1f, x1b);

  // h = relu(x1 @ FF_w^T + FF_b)   (bf16)
  gemm_wmma_kernel<true, true, false><<<gemm_grid(SEQ, DIM), 256, 0, stream>>>(
      x1b, FFwb, hb, FF_b, DIM, DIM, DIM, DIM, 1.0f);
  // adj = h @ FF2_w^T + FF2_b      (f32, reuse tmpf)
  gemm_wmma_kernel<false, false, false><<<gemm_grid(SEQ, DIM), 256, 0, stream>>>(
      hb, FF2wb, tmpf, FF2_b, DIM, DIM, DIM, DIM, 1.0f);
  // out = LN(adj + x1) + decoder
  ln2_out_kernel<<<SEQ, 256, 0, stream>>>(tmpf, x1f, ln2_g, ln2_b, dec,
                                          (float*)d_out);
}
